// Llama3Attention_80650895884775
// MI455X (gfx1250) — matrix operand order repaired and checked by reading, not compiled, not run on hardware
//
#include <hip/hip_runtime.h>
#include <hip/hip_bf16.h>

typedef __attribute__((ext_vector_type(16))) __bf16 v16bf;
typedef __attribute__((ext_vector_type(8)))  __bf16 v8bf;
typedef __attribute__((ext_vector_type(8)))  float  v8f;

#define WMMA_BF16(a, b, c) \
  __builtin_amdgcn_wmma_f32_16x16x32_bf16(false, (a), false, (b), (short)0, (c), false, false)

// Native f32->bf16 (compiler emits v_cvt_*bf16_f32 on gfx1250)
__device__ __forceinline__ __bf16 to_bf16(float f) { return (__bf16)f; }

__device__ __forceinline__ v8f vzero8() {
  v8f v;
#pragma unroll
  for (int e = 0; e < 8; ++e) v[e] = 0.0f;
  return v;
}

// Native sin/cos with range reduction to [-pi, pi] (args reach ~2047 rad).
__device__ __forceinline__ void fast_sincos(float x, float* s, float* c) {
  float r = x * 0.15915494309189535f;       // x / (2*pi)
  r = r - rintf(r);                          // frac in [-0.5, 0.5]
  float xr = r * 6.283185307179586f;         // back to radians, [-pi, pi]
  *s = __builtin_amdgcn_sinf(xr);
  *c = __builtin_amdgcn_cosf(xr);
}

// ---------------------------------------------------------------------------
// Kernel 1: fp32 [K][N] row-major  ->  bf16 [N][K] (transposed) so WMMA
// B-fragments (fixed column, 16 contiguous K) become single 32B loads.
// ---------------------------------------------------------------------------
__global__ void k_transpose_bf16(const float* __restrict__ src,
                                 __bf16* __restrict__ dst, int K, int N) {
  long long idx = (long long)blockIdx.x * blockDim.x + threadIdx.x;
  long long total = (long long)K * N;
  if (idx >= total) return;
  int k = (int)(idx % K);
  int n = (int)(idx / K);
  dst[idx] = to_bf16(src[(long long)k * N + n]);
}

// ---------------------------------------------------------------------------
// Kernel 2: QKV GEMM (fp32 A converted on the fly, bf16 WMMA) + fused RoPE.
// Wave computes a 16(M) x 128(N) tile = a full head column, so RoPE pairs
// (d, d+64) are register-local in accumulators j and j+4.
//   Qb [2048][4096] bf16 (row-major -> A-layout for Q@K^T)
//   Kb [2048][1024] bf16 (row-major -> B-layout for Q@K^T, K-dim=d contiguous)
//   VT [1024][2048] bf16 (transposed -> B-layout for P@V, K-dim=t contiguous)
// ---------------------------------------------------------------------------
__global__ __launch_bounds__(256) void k_qkv_rope(
    const float* __restrict__ hidden,   // [2048][4096]
    const __bf16* __restrict__ wqkvT,   // [6144][4096]
    __bf16* __restrict__ Qb, __bf16* __restrict__ Kb, __bf16* __restrict__ VT) {
  const int lane = threadIdx.x & 31;
  const int wave = threadIdx.x >> 5;
  const int mBase = blockIdx.y * 128 + wave * 16;
  const int nBase = blockIdx.x * 128;
  const int lm = lane & 15;
  const int lh = lane >> 4;
  const int ka = 16 * lh;   // A-fragment per-lane K sub-offset
  const int kb = 16 * lh;   // B-fragment per-lane K sub-offset

  v8f acc[8];
#pragma unroll
  for (int j = 0; j < 8; ++j) acc[j] = vzero8();

  const float* arow = hidden + (long long)(mBase + lm) * 4096;
  for (int k = 0; k < 4096; k += 32) {
    union { v16bf v; v8bf h[2]; } A;
    v8f f0 = *(const v8f*)(arow + k + ka);
    v8f f1 = *(const v8f*)(arow + k + ka + 8);
#pragma unroll
    for (int e = 0; e < 8; ++e) { A.h[0][e] = to_bf16(f0[e]); A.h[1][e] = to_bf16(f1[e]); }
#pragma unroll
    for (int j = 0; j < 8; ++j) {
      v16bf B = *(const v16bf*)(wqkvT + (long long)(nBase + 16 * j + lm) * 4096 + k + kb);
      acc[j] = WMMA_BF16(A.v, B, acc[j]);
    }
  }

  // ln(500000)/64
  const float kLnThetaOver64 = 0.20503693f;

  if (nBase < 5120) {  // Q or K head: apply RoPE
    const bool isQ = nBase < 4096;
    __bf16* outp = isQ ? Qb : Kb;
    const int ld = isQ ? 4096 : 1024;
    const int col = isQ ? nBase : (nBase - 4096);
#pragma unroll
    for (int j = 0; j < 4; ++j) {
      int d = 16 * j + lm;  // rotation pair index, 0..63
      float invf = __expf(-(float)d * kLnThetaOver64);
#pragma unroll
      for (int e = 0; e < 8; ++e) {
        int pos = mBase + e + 8 * lh;
        float s, c;
        fast_sincos((float)pos * invf, &s, &c);
        float x1 = acc[j][e], x2 = acc[j + 4][e];
        outp[(long long)pos * ld + col + d]      = to_bf16(x1 * c - x2 * s);
        outp[(long long)pos * ld + col + d + 64] = to_bf16(x2 * c + x1 * s);
      }
    }
  } else {  // V head: store transposed [d][t]
    const int dBase = nBase - 5120;  // hv*128
#pragma unroll
    for (int j = 0; j < 8; ++j) {
      int d = dBase + 16 * j + lm;
#pragma unroll
      for (int e = 0; e < 8; ++e) {
        int pos = mBase + e + 8 * lh;
        VT[(long long)d * 2048 + pos] = to_bf16(acc[j][e]);
      }
    }
  }
}

// ---------------------------------------------------------------------------
// Kernel 3: flash attention. One wave = (q-head, 16-query tile).
// Online softmax; P goes through a wave-private LDS tile to convert
// C-layout -> A-fragment layout for the P@V WMMAs.
// ---------------------------------------------------------------------------
__global__ __launch_bounds__(256) void k_attn(
    const __bf16* __restrict__ Qb,   // [2048][4096]
    const __bf16* __restrict__ Kb,   // [2048][1024]
    const __bf16* __restrict__ VT,   // [1024][2048]
    __bf16* __restrict__ Ob) {       // [2048][4096]
  __shared__ __bf16 lds_p[8][16 * 32];
  const int lane = threadIdx.x & 31;
  const int wave = threadIdx.x >> 5;
  const int wid = blockIdx.x * 8 + wave;
  const int hq = wid & 31;
  const int qt = wid >> 5;
  const int kvh = hq >> 2;
  const int qBase = qt * 16;
  const int lm = lane & 15;
  const int lh = lane >> 4;
  const int ka = 16 * lh;
  const int kb = 16 * lh;

  // Q fragments: 4 x (16x32) covering head dim 128
  union Frag { v16bf v; v8bf h[2]; };
  Frag aq[4];
  const __bf16* qrow = Qb + (long long)(qBase + lm) * 4096 + hq * 128;
#pragma unroll
  for (int i = 0; i < 4; ++i) {
    aq[i].h[0] = *(const v8bf*)(qrow + 32 * i + ka);
    aq[i].h[1] = *(const v8bf*)(qrow + 32 * i + ka + 8);
  }

  v8f o[8];
#pragma unroll
  for (int j = 0; j < 8; ++j) o[j] = vzero8();
  float rmax[8], rsum[8];
#pragma unroll
  for (int e = 0; e < 8; ++e) { rmax[e] = -3.0e30f; rsum[e] = 0.0f; }

  const float sc = 0.08838834764831845f;  // 1/sqrt(128)
  const __bf16* kcol = Kb + kvh * 128;

  for (int s = 0; s < qBase + 16; s += 32) {
    // S = Q @ K^T for 32 keys (two 16-wide C tiles)
    v8f s0 = vzero8(), s1 = vzero8();
#pragma unroll
    for (int i = 0; i < 4; ++i) {
      v16bf b0 = *(const v16bf*)(kcol + (long long)(s + lm) * 1024 + 32 * i + kb);
      v16bf b1 = *(const v16bf*)(kcol + (long long)(s + 16 + lm) * 1024 + 32 * i + kb);
      s0 = WMMA_BF16(aq[i].v, b0, s0);
      s1 = WMMA_BF16(aq[i].v, b1, s1);
    }

    // scale, causal mask, online softmax, stage P(bf16) into LDS
#pragma unroll
    for (int e = 0; e < 8; ++e) {
      int q = qBase + e + 8 * lh;
      float v0 = s0[e] * sc;
      float v1 = s1[e] * sc;
      if (s + lm > q)      v0 = -3.0e30f;
      if (s + 16 + lm > q) v1 = -3.0e30f;
      float vm = fmaxf(v0, v1);
#pragma unroll
      for (int off = 1; off < 16; off <<= 1) vm = fmaxf(vm, __shfl_xor(vm, off, 32));
      float mnew = fmaxf(rmax[e], vm);
      float corr = __expf(rmax[e] - mnew);
      float p0 = __expf(v0 - mnew);
      float p1 = __expf(v1 - mnew);
      float ps = p0 + p1;
#pragma unroll
      for (int off = 1; off < 16; off <<= 1) ps += __shfl_xor(ps, off, 32);
      rsum[e] = rsum[e] * corr + ps;
      rmax[e] = mnew;
#pragma unroll
      for (int j = 0; j < 8; ++j) o[j][e] *= corr;
      int mloc = e + 8 * lh;
      lds_p[wave][mloc * 32 + lm]      = to_bf16(p0);
      lds_p[wave][mloc * 32 + 16 + lm] = to_bf16(p1);
    }

    // wave-private LDS round trip: C-layout -> A-fragment layout
    asm volatile("s_wait_dscnt 0x0" ::: "memory");
    Frag ap;
    ap.h[0] = *(const v8bf*)&lds_p[wave][lm * 32 + ka];
    ap.h[1] = *(const v8bf*)&lds_p[wave][lm * 32 + 8 + ka];

    // O += P @ V  (V transposed: K-dim = token contiguous)
#pragma unroll
    for (int j = 0; j < 8; ++j) {
      v16bf bv = *(const v16bf*)(VT + (long long)(kvh * 128 + 16 * j + lm) * 2048 + s + kb);
      o[j] = WMMA_BF16(ap.v, bv, o[j]);
    }
  }

  // normalize and write bf16 attention output [t][h*128+d]
#pragma unroll
  for (int j = 0; j < 8; ++j) {
#pragma unroll
    for (int e = 0; e < 8; ++e) {
      int t = qBase + e + 8 * lh;
      Ob[(long long)t * 4096 + hq * 128 + 16 * j + lm] = to_bf16(o[j][e] / rsum[e]);
    }
  }
}

// ---------------------------------------------------------------------------
// Kernel 4: out = attn(bf16) @ w_o  -> fp32. Same tiling as kernel 2.
// ---------------------------------------------------------------------------
__global__ __launch_bounds__(256) void k_out_gemm(
    const __bf16* __restrict__ Ab,   // [2048][4096] bf16
    const __bf16* __restrict__ woT,  // [4096][4096] bf16 (transposed)
    float* __restrict__ out) {       // [2048][4096] fp32
  const int lane = threadIdx.x & 31;
  const int wave = threadIdx.x >> 5;
  const int mBase = blockIdx.y * 128 + wave * 16;
  const int nBase = blockIdx.x * 128;
  const int lm = lane & 15;
  const int lh = lane >> 4;
  const int ka = 16 * lh;
  const int kb = 16 * lh;

  v8f acc[8];
#pragma unroll
  for (int j = 0; j < 8; ++j) acc[j] = vzero8();

  const __bf16* arow = Ab + (long long)(mBase + lm) * 4096;
  for (int k = 0; k < 4096; k += 32) {
    union { v16bf v; v8bf h[2]; } A;
    A.h[0] = *(const v8bf*)(arow + k + ka);
    A.h[1] = *(const v8bf*)(arow + k + ka + 8);
#pragma unroll
    for (int j = 0; j < 8; ++j) {
      v16bf B = *(const v16bf*)(woT + (long long)(nBase + 16 * j + lm) * 4096 + k + kb);
      acc[j] = WMMA_BF16(A.v, B, acc[j]);
    }
  }

#pragma unroll
  for (int j = 0; j < 8; ++j) {
#pragma unroll
    for (int e = 0; e < 8; ++e) {
      out[(long long)(mBase + e + 8 * lh) * 4096 + nBase + 16 * j + lm] = acc[j][e];
    }
  }
}

// ---------------------------------------------------------------------------
// Launch
// ---------------------------------------------------------------------------
extern "C" void kernel_launch(void* const* d_in, const int* in_sizes, int n_in,
                              void* d_out, int out_size, void* d_ws, size_t ws_size,
                              hipStream_t stream) {
  (void)in_sizes; (void)n_in; (void)out_size; (void)ws_size;
  // inputs: 0=positions(int32, == arange(T), folded into kernels), 1=hidden,
  //         2=w_qkv, 3=w_o
  const float* hidden = (const float*)d_in[1];
  const float* w_qkv  = (const float*)d_in[2];
  const float* w_o    = (const float*)d_in[3];

  char* ws = (char*)d_ws;
  __bf16* wqkvT = (__bf16*)(ws + 0);           // 6144*4096*2 = 50331648
  __bf16* woT   = (__bf16*)(ws + 50331648);    // 4096*4096*2 = 33554432
  __bf16* Qb    = (__bf16*)(ws + 83886080);    // 2048*4096*2 = 16777216
  __bf16* Kb    = (__bf16*)(ws + 100663296);   // 2048*1024*2 =  4194304
  __bf16* VT    = (__bf16*)(ws + 104857600);   // 1024*2048*2 =  4194304
  __bf16* Ob    = (__bf16*)(ws + 109051904);   // 2048*4096*2 = 16777216
  // total workspace used: 125829120 bytes (~120 MB)

  k_transpose_bf16<<<(6144 * 4096 + 255) / 256, 256, 0, stream>>>(w_qkv, wqkvT, 4096, 6144);
  k_transpose_bf16<<<(4096 * 4096 + 255) / 256, 256, 0, stream>>>(w_o, woT, 4096, 4096);
  k_qkv_rope<<<dim3(48, 16), 256, 0, stream>>>(hidden, wqkvT, Qb, Kb, VT);
  k_attn<<<512, 256, 0, stream>>>(Qb, Kb, VT, Ob);
  k_out_gemm<<<dim3(32, 16), 256, 0, stream>>>(Ob, woT, (float*)d_out);
}
